// DSSA_65180423685210
// MI455X (gfx1250) — matrix operand order repaired and checked by reading, not compiled, not run on hardware
//
#include <hip/hip_runtime.h>
#include <cstdint>
#include <cstddef>

// ---------------------------------------------------------------------------
// DSSA (BiFormer dynamic sparse attention) for MI455X / gfx1250 (wave32, WMMA)
// GEMMs: v_wmma_f32_16x16x32_bf16 with async global->LDS tile staging
// (global_load_async_to_lds_b128 + s_wait_asynccnt). A-fragment LDS
// gathers are element-contiguous so they merge into ds_load_b128; B gathers stride over k rows.
// ---------------------------------------------------------------------------

typedef __bf16 bf16_t;
typedef __attribute__((ext_vector_type(16))) __bf16 v16bf;
typedef __attribute__((ext_vector_type(8)))  float  v8f;

#define NHD   8      // heads
#define HD    32     // head dim
#define NWIN  7
#define NREG  49
#define SREG  64     // tokens per region
#define TOPKR 4
#define TSTOK 256    // topk*S
#define K2SEL 32     // token-level top-k
#define CCH   256
#define HPIX  56
#define WPIX  56
#define BBAT  8
#define MROWS (BBAT*HPIX*WPIX)   // 25088 pixel rows

// A-fragment K index map for 16-bit A 16x32 (ISA 7.12.2):
// e=0..7 -> k = half*8 + e ; e=8..15 -> k = 16 + half*8 + (e-8)
// (two contiguous 8-element runs per lane -> merges into ds_load_b128)
__device__ __forceinline__ int a_kmap(int e, int half) {
  return ((e >> 3) << 4) + (((e >> 1) & 3) << 1) + (e & 1) + (half << 3);
}

// Async copy of one 16-byte chunk global -> LDS (tracked by ASYNCcnt).
// ldsaddr = low 32 bits of generic LDS pointer (== LDS byte offset, ISA 10.2).
__device__ __forceinline__ void async_copy16(void* lds_ptr, const void* gptr) {
  uint32_t la = (uint32_t)(uintptr_t)lds_ptr;
  uint64_t ga = (uint64_t)(uintptr_t)gptr;
  asm volatile("global_load_async_to_lds_b128 %0, %1, off"
               :: "v"(la), "v"(ga) : "memory");
}
__device__ __forceinline__ void async_wait0() {
  asm volatile("s_wait_asynccnt 0" ::: "memory");
}

// ---------------------------------------------------------------------------
// Kernel 1: weights fp32 -> bf16, TRANSPOSED to K-major (wT[k][o]) so GEMM
// B tiles are row-contiguous for async copy (fragment gathers stride over k rows).
// ---------------------------------------------------------------------------
__global__ void cvt_w_kernel(const float* __restrict__ qkv_w,
                             const float* __restrict__ out_w,
                             bf16_t* __restrict__ wqT, bf16_t* __restrict__ woT) {
  int id = blockIdx.x * blockDim.x + threadIdx.x;
  if (id < 768 * 256) {
    int o = id >> 8, k = id & 255;
    wqT[k * 768 + o] = (bf16_t)qkv_w[id];
  }
  if (id < 256 * 256) {
    int o = id >> 8, k = id & 255;
    woT[k * 256 + o] = (bf16_t)out_w[id];
  }
}

// ---------------------------------------------------------------------------
// Kernel 2: QKV GEMM  out[r,o] = sum_c x[r,c] * W[o,c] + b[o]
// M=25088, K=256, N=768. 64x64 tile per WG, 4 waves, WMMA bf16.
// ---------------------------------------------------------------------------
__global__ __launch_bounds__(128) void qkv_gemm_kernel(
    const float* __restrict__ x, const bf16_t* __restrict__ wbfT,
    const float* __restrict__ bias,
    float* __restrict__ qseq, float* __restrict__ kseq,
    float* __restrict__ vseq, float* __restrict__ vgrid) {
  __shared__ float  Af[64][36];   // 144B row stride: 16B aligned, conflict-free
  __shared__ bf16_t Bs[32][72];   // [k][n], 144B row stride
  const int m0 = blockIdx.x * 64;
  const int n0 = blockIdx.y * 64;
  const int tid = threadIdx.x;
  const int wave = tid >> 5, lane = tid & 31;
  v8f acc[4] = {};

  for (int k0 = 0; k0 < 256; k0 += 32) {
    // A: 64 rows x 32 f32 = 512 16B-chunks
#pragma unroll
    for (int it = 0; it < 4; ++it) {
      int ch = tid + it * 128;
      int r = ch >> 3, q = ch & 7;
      async_copy16(&Af[r][q * 4], x + (size_t)(m0 + r) * 256 + k0 + q * 4);
    }
    // B: 32 rows x 64 bf16 = 256 16B-chunks ; Bs[k][n] = WT[(k0+k)*768 + n0+n]
#pragma unroll
    for (int it = 0; it < 2; ++it) {
      int ch = tid + it * 128;
      int r = ch >> 3, q = ch & 7;
      async_copy16(&Bs[r][q * 8], wbfT + (size_t)(k0 + r) * 768 + n0 + q * 8);
    }
    if (k0 < 224)  // hint next A tile toward L2 (global_prefetch_b8)
      __builtin_prefetch(&x[(size_t)(m0 + (tid >> 1)) * 256 + k0 + 32], 0, 1);
    async_wait0();
    __syncthreads();

    v16bf a;
    const int m = wave * 16 + (lane & 15);
    const int hh = lane >> 4;
#pragma unroll
    for (int e = 0; e < 16; ++e) a[e] = (bf16_t)Af[m][a_kmap(e, hh)];
#pragma unroll
    for (int nt = 0; nt < 4; ++nt) {
      v16bf bfr;  // B[k][n]: n = lane%16, k = a_kmap(element, half)
#pragma unroll
      for (int e = 0; e < 16; ++e) bfr[e] = Bs[a_kmap(e, hh)][nt * 16 + (lane & 15)];
      acc[nt] = __builtin_amdgcn_wmma_f32_16x16x32_bf16(
          false, a, false, bfr, (short)0, acc[nt], false, false);
    }
    __syncthreads();
  }

  // epilogue: C/D layout -> scatter
#pragma unroll
  for (int nt = 0; nt < 4; ++nt) {
#pragma unroll
    for (int i = 0; i < 8; ++i) {
      int ml = wave * 16 + i + 8 * (lane >> 4);
      int nl = nt * 16 + (lane & 15);
      int r = m0 + ml;
      int o = n0 + nl;
      float val = acc[nt][i] + bias[o];
      int which = o >> 8;
      int c = o & 255;
      int b  = r / 3136;
      int p  = r % 3136;
      int hp = p / 56, wp = p % 56;
      int n  = (hp >> 3) * 7 + (wp >> 3);
      int s  = ((hp & 7) << 3) + (wp & 7);
      int mh = c >> 5, d = c & 31;
      size_t si = ((((size_t)b * NHD + mh) * NREG + n) * SREG + s) * HD + d;
      if (which == 0)      qseq[si] = val;
      else if (which == 1) kseq[si] = val;
      else { vseq[si] = val; vgrid[(size_t)r * 256 + c] = val; }
    }
  }
}

// ---------------------------------------------------------------------------
// Kernel 3: region average pooling (== mean over S in seq layout)
// ---------------------------------------------------------------------------
__global__ void pool_kernel(const float* __restrict__ qseq,
                            const float* __restrict__ kseq,
                            float* __restrict__ qr, float* __restrict__ kr) {
  int id = blockIdx.x * blockDim.x + threadIdx.x;
  if (id >= BBAT * NREG * CCH) return;
  int c = id & 255;
  int n = (id >> 8) % NREG;
  int b = (id >> 8) / NREG;
  int mh = c >> 5, d = c & 31;
  size_t base = (((size_t)b * NHD + mh) * NREG + n) * SREG * HD + d;
  float sq = 0.f, sk = 0.f;
  for (int s = 0; s < SREG; ++s) {
    sq += qseq[base + (size_t)s * HD];
    sk += kseq[base + (size_t)s * HD];
  }
  qr[id] = sq * (1.f / 64.f);
  kr[id] = sk * (1.f / 64.f);
}

// ---------------------------------------------------------------------------
// Kernel 4: 49x49 region attention + per-row top-4 region indices
// ---------------------------------------------------------------------------
__global__ __launch_bounds__(64) void route_kernel(const float* __restrict__ qr,
                                                   const float* __restrict__ kr,
                                                   int* __restrict__ idx_r) {
  __shared__ float scores[NREG];
  int b  = blockIdx.x / NREG;
  int nq = blockIdx.x % NREG;
  int j  = threadIdx.x;
  if (j < NREG) {
    const float* qp = qr + ((size_t)b * NREG + nq) * CCH;
    const float* kp = kr + ((size_t)b * NREG + j) * CCH;
    float dot = 0.f;
    for (int c = 0; c < CCH; ++c) dot += qp[c] * kp[c];
    scores[j] = dot;
  }
  __syncthreads();
  if (j == 0) {
    for (int t = 0; t < TOPKR; ++t) {
      float mv = -3.4e38f; int mi = 0;
      for (int c2 = 0; c2 < NREG; ++c2)
        if (scores[c2] > mv) { mv = scores[c2]; mi = c2; }
      idx_r[((size_t)b * NREG + nq) * TOPKR + t] = mi;
      scores[mi] = -3.4e38f;
    }
  }
}

// ---------------------------------------------------------------------------
// Kernel 5: main sparse attention (WMMA QK^T, top-32, softmax, AV)
// ---------------------------------------------------------------------------
__global__ __launch_bounds__(64) void attn_kernel(
    const float* __restrict__ qseq, const float* __restrict__ kseq,
    const float* __restrict__ vseq, const int* __restrict__ idx_r,
    float* __restrict__ y) {
  __shared__ bf16_t qa[32][40];     // 80B row stride
  __shared__ bf16_t kbT[32][264];   // [d][token], 528B row stride
  __shared__ float  sc[32][256];
  __shared__ int    ridx[TOPKR];

  const int id  = blockIdx.x;
  const int hh  = id & 1;
  const int n   = (id >> 1) % NREG;
  const int mh  = ((id >> 1) / NREG) % NHD;
  const int b   = (id >> 1) / (NREG * NHD);
  const int tid = threadIdx.x;
  const int wave = tid >> 5, lane = tid & 31;

  if (tid < TOPKR) ridx[tid] = idx_r[((size_t)b * NREG + n) * TOPKR + tid];
  __syncthreads();

  const size_t khead = ((size_t)b * NHD + mh) * NREG;
  const size_t qbase = (khead + n) * SREG * HD + (size_t)hh * 32 * HD;

  for (int i = tid; i < 32 * 32; i += 64) {
    int r = i >> 5, c = i & 31;
    qa[r][c] = (bf16_t)qseq[qbase + (size_t)r * HD + c];
  }
  // stage K transposed: kbT[d][t]  (global reads stay coalesced over c)
  for (int i = tid; i < 256 * 32; i += 64) {
    int t = i >> 5, c = i & 31;
    int rg = t >> 6, sp = t & 63;
    kbT[c][t] = (bf16_t)kseq[(khead + ridx[rg]) * SREG * HD + (size_t)sp * HD + c];
  }
  __syncthreads();

  {
    v16bf a;
    const int m  = wave * 16 + (lane & 15);
    const int h2 = lane >> 4;
#pragma unroll
    for (int e = 0; e < 16; ++e) a[e] = qa[m][a_kmap(e, h2)];
    for (int nt = 0; nt < 16; ++nt) {
      v16bf bfr;  // B[k][n] = Kg[token n][d=k] : n=lane%16, k=a_kmap(e,half)
#pragma unroll
      for (int e = 0; e < 16; ++e) bfr[e] = kbT[a_kmap(e, h2)][nt * 16 + (lane & 15)];
      v8f av = {};
      av = __builtin_amdgcn_wmma_f32_16x16x32_bf16(
          false, a, false, bfr, (short)0, av, false, false);
#pragma unroll
      for (int i = 0; i < 8; ++i)
        sc[wave * 16 + i + 8 * (lane >> 4)][nt * 16 + (lane & 15)] =
            av[i] * 0.0625f;  // scale = 256^-0.5
    }
  }
  __syncthreads();

  for (int rr = 0; rr < 16; ++rr) {
    const int row = wave * 16 + rr;
    float v[8];
#pragma unroll
    for (int t = 0; t < 8; ++t) v[t] = sc[row][lane * 8 + t];

    float selv = 0.f; int seli = 0;
    for (int j = 0; j < K2SEL; ++j) {
      float mv = -3.4e38f; int mi = 0x7fffffff;
#pragma unroll
      for (int t = 0; t < 8; ++t)
        if (v[t] > mv) { mv = v[t]; mi = lane * 8 + t; }
      for (int off = 16; off > 0; off >>= 1) {
        float ov = __shfl_xor(mv, off, 32);
        int   oi = __shfl_xor(mi, off, 32);
        if (ov > mv || (ov == mv && oi < mi)) { mv = ov; mi = oi; }
      }
      if (lane == j) { selv = mv; seli = mi; }
      if ((mi >> 3) == lane) v[mi & 7] = -3.4e38f;
    }
    float mx = __shfl(selv, 0, 32);
    float p  = __expf(selv - mx);
    float sum = p;
    for (int off = 16; off > 0; off >>= 1) sum += __shfl_xor(sum, off, 32);
    p /= sum;

    float acc = 0.f;
    for (int j = 0; j < K2SEL; ++j) {
      float pj = __shfl(p, j, 32);
      int   gi = __shfl(seli, j, 32);
      int rg = gi >> 6, sp = gi & 63;
      acc += pj * vseq[(khead + ridx[rg]) * SREG * HD + (size_t)sp * HD + lane];
    }
    int s  = hh * 32 + row;
    int hp = (n / 7) * 8 + (s >> 3);
    int wp = (n % 7) * 8 + (s & 7);
    y[(((size_t)b * HPIX + hp) * WPIX + wp) * CCH + mh * HD + lane] = acc;
  }
}

// ---------------------------------------------------------------------------
// Kernel 6: LePE 3x3 depthwise conv on v (NHWC grid), fused add into y
// ---------------------------------------------------------------------------
__global__ void lepe_kernel(const float* __restrict__ vgrid,
                            const float* __restrict__ lw,
                            const float* __restrict__ lb,
                            float* __restrict__ y) {
  size_t id = (size_t)blockIdx.x * blockDim.x + threadIdx.x;
  if (id >= (size_t)MROWS * CCH) return;
  int c = (int)(id & 255);
  size_t r = id >> 8;
  int wp = (int)(r % 56); size_t t = r / 56;
  int hp = (int)(t % 56); int b = (int)(t / 56);
  float acc = lb[c];
#pragma unroll
  for (int i = 0; i < 3; ++i) {
    int h2 = hp + i - 1;
    if (h2 < 0 || h2 >= 56) continue;
#pragma unroll
    for (int j = 0; j < 3; ++j) {
      int w2 = wp + j - 1;
      if (w2 < 0 || w2 >= 56) continue;
      acc += lw[c * 9 + i * 3 + j] *
             vgrid[(((size_t)b * 56 + h2) * 56 + w2) * 256 + c];
    }
  }
  y[id] += acc;
}

// ---------------------------------------------------------------------------
// Kernel 7: output projection GEMM (M=25088, K=256, N=256)
// ---------------------------------------------------------------------------
__global__ __launch_bounds__(128) void out_gemm_kernel(
    const float* __restrict__ yin, const bf16_t* __restrict__ wbfT,
    const float* __restrict__ bias, float* __restrict__ out) {
  __shared__ float  Af[64][36];
  __shared__ bf16_t Bs[32][72];
  const int m0 = blockIdx.x * 64;
  const int n0 = blockIdx.y * 64;
  const int tid = threadIdx.x;
  const int wave = tid >> 5, lane = tid & 31;
  v8f acc[4] = {};

  for (int k0 = 0; k0 < 256; k0 += 32) {
#pragma unroll
    for (int it = 0; it < 4; ++it) {
      int ch = tid + it * 128;
      int r = ch >> 3, q = ch & 7;
      async_copy16(&Af[r][q * 4], yin + (size_t)(m0 + r) * 256 + k0 + q * 4);
    }
#pragma unroll
    for (int it = 0; it < 2; ++it) {
      int ch = tid + it * 128;
      int r = ch >> 3, q = ch & 7;
      async_copy16(&Bs[r][q * 8], wbfT + (size_t)(k0 + r) * 256 + n0 + q * 8);
    }
    async_wait0();
    __syncthreads();

    v16bf a;
    const int m = wave * 16 + (lane & 15);
    const int hh = lane >> 4;
#pragma unroll
    for (int e = 0; e < 16; ++e) a[e] = (bf16_t)Af[m][a_kmap(e, hh)];
#pragma unroll
    for (int nt = 0; nt < 4; ++nt) {
      v16bf bfr;
#pragma unroll
      for (int e = 0; e < 16; ++e) bfr[e] = Bs[a_kmap(e, hh)][nt * 16 + (lane & 15)];
      acc[nt] = __builtin_amdgcn_wmma_f32_16x16x32_bf16(
          false, a, false, bfr, (short)0, acc[nt], false, false);
    }
    __syncthreads();
  }

#pragma unroll
  for (int nt = 0; nt < 4; ++nt) {
#pragma unroll
    for (int i = 0; i < 8; ++i) {
      int ml = wave * 16 + i + 8 * (lane >> 4);
      int nl = nt * 16 + (lane & 15);
      int r = m0 + ml;
      int o = n0 + nl;
      out[(size_t)r * 256 + o] = acc[nt][i] + bias[o];
    }
  }
}

// ---------------------------------------------------------------------------
// Host launcher. Workspace need ~130 MB.
// ---------------------------------------------------------------------------
extern "C" void kernel_launch(void* const* d_in, const int* in_sizes, int n_in,
                              void* d_out, int out_size, void* d_ws, size_t ws_size,
                              hipStream_t stream) {
  const float* x      = (const float*)d_in[0];
  const float* qkv_w  = (const float*)d_in[1];
  const float* qkv_b  = (const float*)d_in[2];
  const float* lepe_w = (const float*)d_in[3];
  const float* lepe_b = (const float*)d_in[4];
  const float* out_w  = (const float*)d_in[5];
  const float* out_b  = (const float*)d_in[6];
  float* out = (float*)d_out;

  const size_t SEQ = (size_t)BBAT * NHD * NREG * SREG * HD;  // 6,422,528 floats
  float* qseq  = (float*)d_ws;
  float* kseq  = qseq + SEQ;
  float* vseq  = kseq + SEQ;
  float* vgrid = vseq + SEQ;
  float* y     = vgrid + SEQ;
  float* qr    = y + SEQ;
  float* kr    = qr + (size_t)BBAT * NREG * CCH;
  int*   idxr  = (int*)(kr + (size_t)BBAT * NREG * CCH);
  bf16_t* wqT  = (bf16_t*)(idxr + (size_t)BBAT * NREG * TOPKR);
  bf16_t* woT  = wqT + 768 * 256;

  cvt_w_kernel<<<(768 * 256 + 255) / 256, 256, 0, stream>>>(qkv_w, out_w, wqT, woT);

  qkv_gemm_kernel<<<dim3(MROWS / 64, 768 / 64), 128, 0, stream>>>(
      x, wqT, qkv_b, qseq, kseq, vseq, vgrid);

  pool_kernel<<<(BBAT * NREG * CCH + 255) / 256, 256, 0, stream>>>(qseq, kseq, qr, kr);

  route_kernel<<<BBAT * NREG, 64, 0, stream>>>(qr, kr, idxr);

  attn_kernel<<<BBAT * NHD * NREG * 2, 64, 0, stream>>>(qseq, kseq, vseq, idxr, y);

  lepe_kernel<<<(int)(((size_t)MROWS * CCH + 255) / 256), 256, 0, stream>>>(
      vgrid, lepe_w, lepe_b, y);

  out_gemm_kernel<<<dim3(MROWS / 64, 256 / 64), 128, 0, stream>>>(y, woT, out_b, out);
}
